// MatchMakerNet_45904610460105
// MI455X (gfx1250) — matrix operand order repaired and checked by reading, not compiled, not run on hardware
//
#include <hip/hip_runtime.h>
#include <math.h>

// ---------------- problem constants ----------------
constexpr int B_SZ  = 8;
constexpr int C_IN  = 64;     // per input tensor
constexpr int CH    = 128;    // concat channels
constexpr int N_PTS = 2048;
constexpr int KNN   = 20;

// ---------------- WMMA types (CDNA5 / gfx1250, wave32) ----------------
typedef __attribute__((ext_vector_type(16))) __bf16 v16bf;
typedef __attribute__((ext_vector_type(8)))  float  v8f;

struct __attribute__((aligned(16))) U128 { unsigned int x[4]; };
struct __attribute__((aligned(16))) F32B { U128 lo, hi; };

__device__ inline unsigned short f2bf(float f) {
  unsigned int u = __float_as_uint(f);
  u += 0x7FFFu + ((u >> 16) & 1u);      // round-to-nearest-even
  return (unsigned short)(u >> 16);
}
__device__ inline float bf2f(unsigned short h) {
  return __uint_as_float(((unsigned int)h) << 16);
}

__device__ inline v8f wmma_bf16(v16bf a, v16bf b, v8f c) {
  // D = A(16x32) * B(32x16) + C, f32 accumulate
  return __builtin_amdgcn_wmma_f32_16x16x32_bf16(false, a, false, b, (short)0, c, false, false);
}

// A-fragment: 16(M)x32(K) bf16. Lane L: M = L&15, hi = L>>4.
// elem e holds K = k0 + 16*hi + e
__device__ inline v16bf load_a_frag(const unsigned short* __restrict__ X, int ldx,
                                    int m0, int k0, int lane) {
  int m  = m0 + (lane & 15);
  int hi = lane >> 4;
  const unsigned short* row = X + (size_t)m * ldx + k0 + 16 * hi;
  F32B t;
  t.lo = *(const U128*)(row);
  t.hi = *(const U128*)(row + 8);
  return __builtin_bit_cast(v16bf, t);
}

// B-fragment: 32(K)x16(N) bf16 from row-major Wsrc[col][k] (ld = ldw).
// Lane L: col = c0 + (L&15), hi = L>>4; elem e holds K = k0 + 16*hi + e.
__device__ inline v16bf load_b_frag(const unsigned short* __restrict__ Wsrc, int ldw,
                                    int c0, int k0, int lane) {
  int col = c0 + (lane & 15);
  int hi  = lane >> 4;
  const unsigned short* p = Wsrc + (size_t)col * ldw + k0 + 16 * hi;
  F32B t;
  t.lo = *(const U128*)(p);
  t.hi = *(const U128*)(p + 8);
  return __builtin_bit_cast(v16bf, t);
}

// B-fragment from a per-lane base pointer (bp = row + 16*hi); k0 becomes an
// immediate instruction offset when the caller unrolls.
__device__ inline v16bf load_b_from(const unsigned short* __restrict__ bp, int k0) {
  F32B t;
  t.lo = *(const U128*)(bp + k0);
  t.hi = *(const U128*)(bp + k0 + 8);
  return __builtin_bit_cast(v16bf, t);
}

// ---------------- kernel 1: concat + f32->bf16 pack, point-major (B*N, 128) ----
__global__ void k_pack(const float* __restrict__ f1, const float* __restrict__ f2,
                       unsigned short* __restrict__ X) {
  size_t e = (size_t)blockIdx.x * blockDim.x + threadIdx.x;  // over B*128*N
  if (e >= (size_t)B_SZ * CH * N_PTS) return;
  int n = (int)(e % N_PTS);
  int c = (int)((e / N_PTS) % CH);
  int b = (int)(e / ((size_t)N_PTS * CH));
  float v = (c < C_IN) ? f1[((size_t)b * C_IN + c) * N_PTS + n]
                       : f2[((size_t)b * C_IN + (c - C_IN)) * N_PTS + n];
  X[((size_t)b * N_PTS + n) * CH + c] = f2bf(v);
}

// ---------------- kernel 2: squared norms (consistent with bf16-rounded X) ----
__global__ void k_xx(const float* __restrict__ f1, const float* __restrict__ f2,
                     float* __restrict__ xx) {
  int n = blockIdx.x * blockDim.x + threadIdx.x;
  int b = blockIdx.y;
  if (n >= N_PTS) return;
  float s = 0.f;
  for (int c = 0; c < C_IN; ++c) {
    float a = bf2f(f2bf(f1[((size_t)b * C_IN + c) * N_PTS + n]));
    float d = bf2f(f2bf(f2[((size_t)b * C_IN + c) * N_PTS + n]));
    s += a * a + d * d;
  }
  xx[(size_t)b * N_PTS + n] = s;
}

// ---------------- kernel 3: weight prep (bf16 + algebraic refactor) ----------
__global__ void k_wprep(const float* __restrict__ ca, const float* __restrict__ cb,
                        const float* __restrict__ cs, const float* __restrict__ c1,
                        unsigned short* __restrict__ WdA, unsigned short* __restrict__ WcdA,
                        unsigned short* __restrict__ WdB, unsigned short* __restrict__ WcdB,
                        unsigned short* __restrict__ Wcomb) {
  const int T0 = 128 * 128, T1 = T0 + 64 * 128, T2 = T1 + 128 * 192;
  for (int e = blockIdx.x * blockDim.x + threadIdx.x; e < T2; e += gridDim.x * blockDim.x) {
    if (e < T0) {
      int o = e / 128, c = e % 128;
      WdA[e]  = f2bf(ca[o * 256 + c]);
      WcdA[e] = f2bf(ca[o * 256 + 128 + c] - ca[o * 256 + c]);
    } else if (e < T1) {
      int i = e - T0, o = i / 128, c = i % 128;
      WdB[i]  = f2bf(cb[o * 256 + c]);
      WcdB[i] = f2bf(cb[o * 256 + 128 + c] - cb[o * 256 + c]);
    } else {
      int i = e - T1, o = i / 192, c = i % 192;
      float v = c1[o * 192 + c] + ((c < 128) ? cs[o * 128 + c] : 0.f);
      Wcomb[i] = f2bf(v);
    }
  }
}

// ---------------- kernel 4: fused kNN (WMMA GEMM + streaming top-20) --------
// One wave per 32 rows; software-pipelined: B fragments for column tile t+1
// are issued before the WMMAs of tile t, per-lane running pointers (no muls).
__global__ __launch_bounds__(32) void k_knn(const unsigned short* __restrict__ X, int ldx,
                                            const float* __restrict__ xx,
                                            int* __restrict__ idxout) {
  __shared__ float sc[32 * 16];
  __shared__ float lv[32 * KNN];
  __shared__ int   li[32 * KNN];

  const int lane = threadIdx.x;
  const int b    = blockIdx.y;
  const int m0   = blockIdx.x * 32;
  const int hi   = lane >> 4;

  const unsigned short* Xb  = X + (size_t)b * N_PTS * ldx;
  const float*          xxb = xx + (size_t)b * N_PTS;
  int*                  ib  = idxout + (size_t)b * N_PTS * KNN;

  // Preload A fragments for our 32 rows (2 row-tiles x 4 K-chunks)
  v16bf a[2][4];
#pragma unroll
  for (int t = 0; t < 2; ++t)
#pragma unroll
    for (int kc = 0; kc < 4; ++kc)
      a[t][kc] = load_a_frag(Xb, ldx, m0 + 16 * t, 32 * kc, lane);

  float xm[2][8];
#pragma unroll
  for (int t = 0; t < 2; ++t)
#pragma unroll
    for (int r = 0; r < 8; ++r)
      xm[t][r] = xxb[m0 + 16 * t + r + 8 * hi];

  int count = 0;
  float thr = -1e30f;

  // per-lane running pointers for B fragments / column norms
  const unsigned short* bp = Xb + (size_t)(lane & 15) * ldx + 16 * hi;
  const float*          xp = xxb + (lane & 15);
  const size_t tilestep = (size_t)16 * ldx;

  v16bf bcur[4], bnxt[4];
#pragma unroll
  for (int kc = 0; kc < 4; ++kc) bcur[kc] = load_b_from(bp, 32 * kc);
  float xn_cur = *xp;

  for (int n0 = 0; n0 < N_PTS; n0 += 16) {
    // prefetch next tile's operands while this tile computes
    const unsigned short* bpn = bp + tilestep;
    float xn_nxt = 0.f;
    if (n0 + 16 < N_PTS) {
#pragma unroll
      for (int kc = 0; kc < 4; ++kc) bnxt[kc] = load_b_from(bpn, 32 * kc);
      xn_nxt = xp[16];
    }

    v8f c0 = {};
    v8f c1 = {};
#pragma unroll
    for (int kc = 0; kc < 4; ++kc) {
      c0 = wmma_bf16(a[0][kc], bcur[kc], c0);
      c1 = wmma_bf16(a[1][kc], bcur[kc], c1);
    }

#pragma unroll
    for (int r = 0; r < 8; ++r) {
      sc[(r + 8 * hi) * 16 + (lane & 15)]      = 2.f * c0[r] - xm[0][r] - xn_cur;
      sc[(16 + r + 8 * hi) * 16 + (lane & 15)] = 2.f * c1[r] - xm[1][r] - xn_cur;
    }
    __syncthreads();

    // each lane owns one row; merge 16 candidates into its LDS top-20 list
    float* rv = &lv[lane * KNN];
    int*   ri = &li[lane * KNN];
    for (int j = 0; j < 16; ++j) {
      float cand = sc[lane * 16 + j];
      int   ci   = n0 + j;
      if (count < KNN || cand > thr) {
        int pos = (count < KNN) ? count : (KNN - 1);
        while (pos > 0 && rv[pos - 1] < cand) {
          rv[pos] = rv[pos - 1];
          ri[pos] = ri[pos - 1];
          --pos;
        }
        rv[pos] = cand;
        ri[pos] = ci;
        if (count < KNN) ++count;
        thr = rv[count - 1];
      }
    }
    __syncthreads();

#pragma unroll
    for (int kc = 0; kc < 4; ++kc) bcur[kc] = bnxt[kc];
    xn_cur = xn_nxt;
    bp = bpn;
    xp += 16;
  }

  const int m = m0 + lane;
  for (int j = 0; j < KNN; ++j) ib[(size_t)m * KNN + j] = li[lane * KNN + j];
}

// ---------------- kernel 5: WMMA GEMM  D(M x O) = X(M x K) * W(O x K)^T ------
// One wave per 16 rows; A fragments loaded once, reused across all O column
// tiles. NK = K/32 as template parameter keeps fragment arrays in registers.
template <int NK>
__global__ __launch_bounds__(32) void k_gemm(const unsigned short* __restrict__ X, int ldx,
                                             const unsigned short* __restrict__ W, int Odim,
                                             unsigned short* __restrict__ Obf,
                                             float* __restrict__ Of32,
                                             int ldo, int ocol0) {
  const int Kdim = NK * 32;
  const int lane = threadIdx.x;
  const int m0   = blockIdx.x * 16;
  const int hi   = lane >> 4;

  v16bf a[NK];
#pragma unroll
  for (int kc = 0; kc < NK; ++kc) a[kc] = load_a_frag(X, ldx, m0, 32 * kc, lane);

  for (int c0 = 0; c0 < Odim; c0 += 16) {
    v8f acc = {};
#pragma unroll
    for (int kc = 0; kc < NK; ++kc) {
      v16bf bv = load_b_frag(W, Kdim, c0, 32 * kc, lane);
      acc = wmma_bf16(a[kc], bv, acc);
    }
    const int col = c0 + (lane & 15);
#pragma unroll
    for (int r = 0; r < 8; ++r) {
      int   m = m0 + r + 8 * hi;
      float v = acc[r];
      if (Of32) Of32[(size_t)m * ldo + ocol0 + col] = v;
      else      Obf [(size_t)m * ldo + ocol0 + col] = f2bf(v);
    }
  }
}

// ---------------- kernel 6: gather + max_k + const-term add (edge conv) ------
__global__ void k_gathermax(const unsigned short* __restrict__ Y, int ldy,
                            const unsigned short* __restrict__ Z, int ldz,
                            const int* __restrict__ idx,
                            unsigned short* __restrict__ Out, int ldo, int ocol0,
                            float* __restrict__ xxout) {
  __shared__ float red[128];
  const int b = blockIdx.y, n = blockIdx.x, o = threadIdx.x;
  const unsigned short* Yb = Y + (size_t)b * N_PTS * ldy;
  const int* ib = idx + ((size_t)b * N_PTS + n) * KNN;

  int jn = ib[0];
  float m = -1e30f;
  for (int k = 0; k < KNN; ++k) {
    int j = jn;
    if (k + 1 < KNN) {
      jn = ib[k + 1];
      __builtin_prefetch(Yb + (size_t)jn * ldy + o, 0, 1);  // global_prefetch
    }
    m = fmaxf(m, bf2f(Yb[(size_t)j * ldy + o]));
  }
  float z   = bf2f(Z[((size_t)b * N_PTS + n) * ldz + o]);
  float res = m + z;
  Out[((size_t)b * N_PTS + n) * ldo + ocol0 + o] = f2bf(res);

  if (xxout) {
    float rb = bf2f(f2bf(res));
    red[o] = rb * rb;
    __syncthreads();
    for (int s = blockDim.x >> 1; s > 0; s >>= 1) {
      if (o < s) red[o] += red[o + s];
      __syncthreads();
    }
    if (o == 0) xxout[(size_t)b * N_PTS + n] = red[0];
  }
}

// ---------------- kernel 7: max + mean pooling over N ------------------------
__global__ __launch_bounds__(128) void k_pool(const float* __restrict__ outc,
                                              float* __restrict__ feat) {
  const int b = blockIdx.x, o = threadIdx.x;
  const float* p = outc + (size_t)b * N_PTS * CH + o;
  float mx = -1e30f, sm = 0.f;
  for (int n = 0; n < N_PTS; ++n) {
    float v = p[(size_t)n * CH];
    mx = fmaxf(mx, v);
    sm += v;
  }
  feat[b * 256 + o]       = mx;
  feat[b * 256 + 128 + o] = sm * (1.f / N_PTS);
}

// ---------------- kernel 8: fc1 + BN + LeakyReLU + fc2 ----------------------
__global__ __launch_bounds__(64) void k_head(const float* __restrict__ feat,
                                             const float* __restrict__ fc1w,
                                             const float* __restrict__ fc1b,
                                             const float* __restrict__ gam,
                                             const float* __restrict__ bet,
                                             const float* __restrict__ mn,
                                             const float* __restrict__ vr,
                                             const float* __restrict__ fc2w,
                                             const float* __restrict__ fc2b,
                                             float* __restrict__ out) {
  __shared__ float red[64];
  const int b = blockIdx.x, o = threadIdx.x;
  float s = fc1b[o];
  for (int c = 0; c < 256; ++c) s += feat[b * 256 + c] * fc1w[o * 256 + c];
  s = (s - mn[o]) * rsqrtf(vr[o] + 1e-5f) * gam[o] + bet[o];
  s = (s >= 0.f) ? s : 0.01f * s;
  red[o] = s * fc2w[o];
  __syncthreads();
  for (int st = 32; st > 0; st >>= 1) {
    if (o < st) red[o] += red[o + st];
    __syncthreads();
  }
  if (o == 0) out[b] = red[0] + fc2b[0];
}

// ---------------- host-side orchestration -----------------------------------
extern "C" void kernel_launch(void* const* d_in, const int* in_sizes, int n_in,
                              void* d_out, int out_size, void* d_ws, size_t ws_size,
                              hipStream_t stream) {
  const float* f1   = (const float*)d_in[0];
  const float* f2   = (const float*)d_in[1];
  const float* caw  = (const float*)d_in[2];   // (128,256)
  const float* cbw  = (const float*)d_in[3];   // (64,256)
  const float* csw  = (const float*)d_in[4];   // (128,128)
  const float* c1w  = (const float*)d_in[5];   // (128,192)
  const float* fc1w = (const float*)d_in[6];
  const float* fc1b = (const float*)d_in[7];
  const float* bng  = (const float*)d_in[8];
  const float* bnb  = (const float*)d_in[9];
  const float* bnm  = (const float*)d_in[10];
  const float* bnv  = (const float*)d_in[11];
  const float* fc2w = (const float*)d_in[12];
  const float* fc2b = (const float*)d_in[13];
  float* out = (float*)d_out;

  char* ws = (char*)d_ws;
  size_t off = 0;
  auto carve = [&](size_t bytes) { char* p = ws + off; off += (bytes + 255) & ~(size_t)255; return p; };

  unsigned short* Xbf   = (unsigned short*)carve((size_t)B_SZ * N_PTS * CH * 2);   // 4 MB
  float*          xx0   = (float*)carve((size_t)B_SZ * N_PTS * 4);
  float*          xxa   = (float*)carve((size_t)B_SZ * N_PTS * 4);
  int*            idxb  = (int*)carve((size_t)B_SZ * N_PTS * KNN * 4);             // 1.25 MB (reused)
  unsigned short* WdA   = (unsigned short*)carve(128 * 128 * 2);
  unsigned short* WcdA  = (unsigned short*)carve(128 * 128 * 2);
  unsigned short* WdB   = (unsigned short*)carve(64 * 128 * 2);
  unsigned short* WcdB  = (unsigned short*)carve(64 * 128 * 2);
  unsigned short* Wcomb = (unsigned short*)carve(128 * 192 * 2);
  unsigned short* Ya    = (unsigned short*)carve((size_t)B_SZ * N_PTS * CH * 2);   // reused for Yb
  unsigned short* Za    = (unsigned short*)carve((size_t)B_SZ * N_PTS * CH * 2);   // reused for Zb
  unsigned short* cat   = (unsigned short*)carve((size_t)B_SZ * N_PTS * 192 * 2);  // 6 MB
  float*          outc  = (float*)carve((size_t)B_SZ * N_PTS * CH * 4);            // 8 MB
  float*          feat  = (float*)carve((size_t)B_SZ * 256 * 4);
  (void)ws_size; (void)in_sizes; (void)n_in; (void)out_size;

  const size_t MROWS = (size_t)B_SZ * N_PTS;          // 16384
  // 1) pack concat(f1,f2) -> bf16 point-major
  k_pack<<<(unsigned)((MROWS * CH + 255) / 256), 256, 0, stream>>>(f1, f2, Xbf);
  // 2) squared norms
  k_xx<<<dim3(N_PTS / 256, B_SZ), 256, 0, stream>>>(f1, f2, xx0);
  // 3) weight prep (bf16, edge-conv refactor, conv1+skip fusion)
  k_wprep<<<64, 256, 0, stream>>>(caw, cbw, csw, c1w, WdA, WcdA, WdB, WcdB, Wcomb);
  // 4) kNN #1 on concat features
  k_knn<<<dim3(N_PTS / 32, B_SZ), 32, 0, stream>>>(Xbf, CH, xx0, idxb);
  // 5) Ya = X*WdA^T, Za = X*(WcA-WdA)^T   (each 16384x128x128)
  k_gemm<4><<<(unsigned)(MROWS / 16), 32, 0, stream>>>(Xbf, CH, WdA,  128, Ya, nullptr, CH, 0);
  k_gemm<4><<<(unsigned)(MROWS / 16), 32, 0, stream>>>(Xbf, CH, WcdA, 128, Za, nullptr, CH, 0);
  // 6) out_a = max_k Ya[idx] + Za  -> cat[:, :128]; also ||out_a||^2 for kNN #2
  k_gathermax<<<dim3(N_PTS, B_SZ), 128, 0, stream>>>(Ya, CH, Za, CH, idxb, cat, 192, 0, xxa);
  // 7) kNN #2 on out_a (rows of cat, stride 192)
  k_knn<<<dim3(N_PTS / 32, B_SZ), 32, 0, stream>>>(cat, 192, xxa, idxb);
  // 8) Yb = out_a*WdB^T, Zb = out_a*(WcB-WdB)^T  (16384x64x128), reuse Ya/Za
  k_gemm<4><<<(unsigned)(MROWS / 16), 32, 0, stream>>>(cat, 192, WdB,  64, Ya, nullptr, 64, 0);
  k_gemm<4><<<(unsigned)(MROWS / 16), 32, 0, stream>>>(cat, 192, WcdB, 64, Za, nullptr, 64, 0);
  // 9) out_b = max_k Yb[idx] + Zb -> cat[:, 128:192]
  k_gathermax<<<dim3(N_PTS, B_SZ), 64, 0, stream>>>(Ya, 64, Za, 64, idxb, cat, 192, 128, nullptr);
  // 10) outc = cat(192) * Wcomb^T  (f32 output for exact pooling)
  k_gemm<6><<<(unsigned)(MROWS / 16), 32, 0, stream>>>(cat, 192, Wcomb, 128, nullptr, outc, CH, 0);
  // 11) global max+mean pool
  k_pool<<<B_SZ, 128, 0, stream>>>(outc, feat);
  // 12) fc1 + BN + LeakyReLU + fc2
  k_head<<<B_SZ, 64, 0, stream>>>(feat, fc1w, fc1b, bng, bnb, bnm, bnv, fc2w, fc2b, out);
}
